// SpikingNeuralNetwork_14877766713532
// MI455X (gfx1250) — matrix operand order repaired and checked by reading, not compiled, not run on hardware
//
#include <hip/hip_runtime.h>
#include <hip/hip_bf16.h>
#include <stdint.h>
#include <stddef.h>

typedef __attribute__((ext_vector_type(16))) __bf16 v16bf;
typedef __attribute__((ext_vector_type(8)))  __bf16 v8bf;
typedef __attribute__((ext_vector_type(8)))  float  v8f;

static constexpr int B = 64, T = 512, I = 256, H = 512, O = 128;
static constexpr float DECAY = 0.9f, THR = 1.0f;

// output tuple layout (flat f32): o_spk [B,T,O], h_spk [B,T,H], o_mem [B,T,O], h_mem [B,T,H]
static constexpr size_t OFF_OSPK = 0;
static constexpr size_t OFF_HSPK = (size_t)B * T * O;
static constexpr size_t OFF_OMEM = OFF_HSPK + (size_t)B * T * H;
static constexpr size_t OFF_HMEM = OFF_OMEM + (size_t)B * T * O;

__device__ __forceinline__ unsigned short f2bf(float f) {
  union { float f; unsigned u; } v; v.f = f;
  unsigned u = v.u;
  u += 0x7FFFu + ((u >> 16) & 1u);   // round to nearest even
  return (unsigned short)(u >> 16);
}

__global__ void cvt_f32_bf16(const float* __restrict__ src,
                             unsigned short* __restrict__ dst, int n) {
  int i = blockIdx.x * blockDim.x + threadIdx.x;
  int stride = gridDim.x * blockDim.x;
  for (; i < n; i += stride) dst[i] = f2bf(src[i]);
}

// A fragment (16x32 bf16, MxK): lane&15 = M row; elems 0..7 <-> K=kb..kb+7,
// elems 8..15 <-> K=16+kb..16+kb+7, kb = (lane>>4)*8.  ldk = K-stride in elems.
__device__ __forceinline__ v16bf load_frag_a(const __bf16* base, int ldk,
                                             int row, int k0, int lane) {
  int kb = k0 + ((lane >> 4) << 3);
  const v8bf* p0 = (const v8bf*)(base + (size_t)row * ldk + kb);
  const v8bf* p1 = (const v8bf*)(base + (size_t)row * ldk + kb + 16);
  v8bf lo = *p0, hi = *p1;
  v16bf a;
#pragma unroll
  for (int i = 0; i < 8; ++i) { a[i] = lo[i]; a[i + 8] = hi[i]; }
  return a;
}

// B fragment (32x16 bf16, KxN) where memory holds W[n][k] row-major:
// lane&15 = N; elems 0..7 <-> K=kb..kb+7, elems 8..15 <-> K=16+kb..16+kb+7, kb = (lane>>4)*8.
__device__ __forceinline__ v16bf load_frag_b(const __bf16* base, int ldk,
                                             int nrow, int k0, int lane) {
  int kb = k0 + ((lane >> 4) << 3);
  const v8bf* p0 = (const v8bf*)(base + (size_t)nrow * ldk + kb);
  const v8bf* p1 = (const v8bf*)(base + (size_t)nrow * ldk + kb + 16);
  v8bf lo = *p0, hi = *p1;
  v16bf b;
#pragma unroll
  for (int i = 0; i < 8; ++i) { b[i] = lo[i]; b[i + 8] = hi[i]; }
  return b;
}

// Async copy 2x16B (32B) from global to LDS, tracked by ASYNCcnt.
// ldsoff: byte offset of destination within the workgroup LDS segment.
// voff:   per-thread unsigned byte offset added to the SGPR base (GVS mode).
__device__ __forceinline__ void async_copy32(unsigned ldsoff, const void* gbase, int voff) {
  asm volatile("global_load_async_to_lds_b128 %0, %1, %2\n\t"
               "global_load_async_to_lds_b128 %0, %1, %2 offset:16"
               :: "v"(ldsoff), "v"(voff), "s"(gbase)
               : "memory");
}

__device__ __forceinline__ void wait_asynccnt_le2() {
  asm volatile("s_wait_asynccnt 0x2" ::: "memory");
}

// hin[t][b][h] = x[b,t,:] @ Wi[h,:] + bi[h].  M=B*T rows of x, N=H, K=I.
__global__ __launch_bounds__(256) void hin_gemm(
    const unsigned short* __restrict__ xbf, const unsigned short* __restrict__ wibf,
    const float* __restrict__ bi, float* __restrict__ hin) {
  const int lane = threadIdx.x & 31;
  const int wave = threadIdx.x >> 5;               // 8 waves / block
  const int tile = blockIdx.x * 8 + wave;          // 65536 tiles total
  const int nTiles = H / 16;                       // 32
  const int mTile = tile / nTiles;
  const int nTile = tile % nTiles;
  const __bf16* A  = (const __bf16*)xbf;           // [B*T][I]
  const __bf16* Bw = (const __bf16*)wibf;          // [H][I]
  const int arow = mTile * 16 + (lane & 15);
  const int brow = nTile * 16 + (lane & 15);
  v8f c = {};
#pragma unroll
  for (int k0 = 0; k0 < I; k0 += 32) {
    v16bf a = load_frag_a(A, I, arow, k0, lane);
    v16bf b = load_frag_b(Bw, I, brow, k0, lane);
    c = __builtin_amdgcn_wmma_f32_16x16x32_bf16(false, a, false, b,
                                                (short)0, c, false, false);
  }
  const int n = nTile * 16 + (lane & 15);
  const float bias = bi[n];
#pragma unroll
  for (int v = 0; v < 8; ++v) {
    int m = v + ((lane >> 4) << 3);
    int r = mTile * 16 + m;                        // r = b*T + t
    int bb = r / T, tt = r % T;
    hin[((size_t)tt * B + bb) * H + n] = c[v] + bias;   // store as [T][B][H]
  }
}

// Sequential LIF scan.  4 blocks x 1024 threads (32 waves). Block owns 16 batch
// rows; wave w owns h-columns [16w,16w+16) with h_mem resident in its WMMA
// accumulator; waves 0..7 also own o-columns [16w,16w+16) (o_mem resident).
// hin slabs (32KB/step) are double-buffered in LDS via async global->LDS copies
// issued one step ahead so HBM latency hides behind the recurrent GEMM.
__global__ __launch_bounds__(1024) void snn_scan(
    const unsigned short* __restrict__ whbf, const unsigned short* __restrict__ wobf,
    const float* __restrict__ bh, const float* __restrict__ bo,
    const float* __restrict__ hin, float* __restrict__ out) {
  constexpr int SP = H + 8;                        // padded spike stride (bf16 elems)
  constexpr int HP = H + 4;                        // padded hin stride (f32 elems)
  __shared__ unsigned short spk_lds[16 * SP];      // 16.25 KB
  __shared__ float hin_lds[2][16 * HP];            // 64.5 KB (double buffer)

  const int tid   = threadIdx.x;
  const int lane  = tid & 31;
  const int wave  = tid >> 5;                      // 0..31
  const int bBase = blockIdx.x * 16;
  const int nCol  = wave * 16 + (lane & 15);       // this wave's h column
  const float bh_n = bh[nCol];
  const int mRowA = lane & 15;                     // A-fragment row (batch-local)
  const __bf16* Wh   = (const __bf16*)whbf;        // [H][H]
  const __bf16* Wo   = (const __bf16*)wobf;        // [O][H]
  const __bf16* spkA = (const __bf16*)spk_lds;

  const bool doO  = wave < (O / 16);               // waves 0..7
  const int  oCol = wave * 16 + (lane & 15);
  const float bo_n = doO ? bo[oCol] : 0.0f;

  // async-copy addressing: each thread moves 32B; 64 threads cover one 2KB row
  const int cpRow   = tid >> 6;                    // batch-local row 0..15
  const int cpVoff  = tid * 32;                    // linear byte offset in slab
  const unsigned cpLds0 = (unsigned)(uintptr_t)&hin_lds[0][0]
                        + (unsigned)(cpRow * HP * 4 + (tid & 63) * 32);
  const unsigned cpLds1 = (unsigned)(uintptr_t)&hin_lds[1][0]
                        + (unsigned)(cpRow * HP * 4 + (tid & 63) * 32);

  v8f hmem = {};
  v8f omem = {};

  // prefetch slab for t=0 into buffer 0
  async_copy32(cpLds0, hin + ((size_t)0 * B + bBase) * H, cpVoff);

  for (int t = 0; t < T; ++t) {
    // issue async prefetch of step t+1's slab into the other buffer
    const int tn = (t + 1 < T) ? (t + 1) : 0;      // wrap harmlessly at the end
    async_copy32(((t + 1) & 1) ? cpLds1 : cpLds0,
                 hin + ((size_t)tn * B + bBase) * H, cpVoff);

    // (a) recurrent GEMM: rec = prev_spk @ Wh.T  (skip at t=0)
    v8f rec = {};
    if (t > 0) {
#pragma unroll
      for (int k0 = 0; k0 < H; k0 += 32) {
        v16bf a = load_frag_a(spkA, SP, mRowA, k0, lane);
        v16bf b = load_frag_b(Wh, H, nCol, k0, lane);
        rec = __builtin_amdgcn_wmma_f32_16x16x32_bf16(false, a, false, b,
                                                      (short)0, rec, false, false);
      }
    }
    // my 2 copies for buf[t&1] retired (in-order); 2 outstanding are for t+1
    wait_asynccnt_le2();
    __syncthreads();   // prev-spike reads done + slab t visible group-wide

    // (b) hidden LIF; write new spikes to LDS + outputs
    const float* drv = &hin_lds[t & 1][0];
#pragma unroll
    for (int v = 0; v < 8; ++v) {
      int m = v + ((lane >> 4) << 3);              // batch-local row 0..15
      int bb = bBase + m;
      float drive = drv[m * HP + nCol];
      float inp = drive + ((t > 0) ? (rec[v] + bh_n) : 0.0f);
      float mem = DECAY * hmem[v] + inp;
      float spk = (mem >= THR) ? 1.0f : 0.0f;
      mem = (spk > 0.0f) ? 0.0f : mem;
      hmem[v] = mem;
      spk_lds[m * SP + nCol] = f2bf(spk);
      size_t oi = ((size_t)bb * T + t) * H + nCol;
      out[OFF_HSPK + oi] = spk;
      out[OFF_HMEM + oi] = mem;
    }
    __syncthreads();   // new spikes visible for output GEMM / next step

    // (c) output GEMM + LIF (waves 0..7)
    if (doO) {
      v8f oc = {};
#pragma unroll
      for (int k0 = 0; k0 < H; k0 += 32) {
        v16bf a = load_frag_a(spkA, SP, mRowA, k0, lane);
        v16bf b = load_frag_b(Wo, H, oCol, k0, lane);
        oc = __builtin_amdgcn_wmma_f32_16x16x32_bf16(false, a, false, b,
                                                     (short)0, oc, false, false);
      }
#pragma unroll
      for (int v = 0; v < 8; ++v) {
        int m = v + ((lane >> 4) << 3);
        int bb = bBase + m;
        float mem = DECAY * omem[v] + (oc[v] + bo_n);
        float spk = (mem >= THR) ? 1.0f : 0.0f;
        mem = (spk > 0.0f) ? 0.0f : mem;
        omem[v] = mem;
        size_t oi = ((size_t)bb * T + t) * O + oCol;
        out[OFF_OSPK + oi] = spk;
        out[OFF_OMEM + oi] = mem;
      }
    }
  }
}

extern "C" void kernel_launch(void* const* d_in, const int* in_sizes, int n_in,
                              void* d_out, int out_size, void* d_ws, size_t ws_size,
                              hipStream_t stream) {
  const float* x  = (const float*)d_in[0];
  const float* Wi = (const float*)d_in[1];
  const float* bi = (const float*)d_in[2];
  const float* Wh = (const float*)d_in[3];
  const float* bh = (const float*)d_in[4];
  const float* Wo = (const float*)d_in[5];
  const float* bo = (const float*)d_in[6];
  float* out = (float*)d_out;

  // workspace layout (all offsets 256B-aligned by construction)
  char* ws = (char*)d_ws;
  size_t off = 0;
  unsigned short* xbf  = (unsigned short*)(ws + off); off += (size_t)B * T * I * 2; // 16 MiB
  unsigned short* wibf = (unsigned short*)(ws + off); off += (size_t)H * I * 2;
  unsigned short* whbf = (unsigned short*)(ws + off); off += (size_t)H * H * 2;
  unsigned short* wobf = (unsigned short*)(ws + off); off += (size_t)O * H * 2;
  float*          hin  = (float*)(ws + off);          off += (size_t)T * B * H * 4; // 64 MiB
  (void)ws_size; (void)in_sizes; (void)n_in; (void)out_size;

  // f32 -> bf16 conversions
  cvt_f32_bf16<<<(B * T * I + 255) / 256, 256, 0, stream>>>(x,  xbf,  B * T * I);
  cvt_f32_bf16<<<(H * I + 255) / 256,     256, 0, stream>>>(Wi, wibf, H * I);
  cvt_f32_bf16<<<(H * H + 255) / 256,     256, 0, stream>>>(Wh, whbf, H * H);
  cvt_f32_bf16<<<(O * H + 255) / 256,     256, 0, stream>>>(Wo, wobf, O * H);

  // feed-forward drive: 65536 16x16 tiles, 8 waves/block
  hin_gemm<<<8192, 256, 0, stream>>>(xbf, wibf, bi, hin);

  // sequential scan: 4 blocks (16 batch rows each), 32 waves/block
  snn_scan<<<4, 1024, 0, stream>>>(whbf, wobf, bh, bo, hin, out);
}
